// ChempropLayer_1760936591674
// MI455X (gfx1250) — matrix operand order repaired and checked by reading, not compiled, not run on hardware
//
#include <hip/hip_runtime.h>

#define N_NODES 50000
#define N_EDGES 300000
#define HIDDEN  256

typedef __attribute__((ext_vector_type(16))) __bf16 v16bf;
typedef __attribute__((ext_vector_type(8)))  float  v8f;

__device__ __forceinline__ unsigned short f32_to_bf16(float f) {
    union { float f; unsigned int u; } c; c.f = f;
    unsigned int u = c.u;
    u += 0x7FFFu + ((u >> 16) & 1u);   // round to nearest even
    return (unsigned short)(u >> 16);
}

// ---------------------------------------------------------------- zero M_v
__global__ void k_zero(float4* __restrict__ p, int n4) {
    int i = blockIdx.x * blockDim.x + threadIdx.x;
    if (i < n4) p[i] = make_float4(0.f, 0.f, 0.f, 0.f);
}

// ------------------------------------------- W -> bf16 WMMA-B-fragment order
// Fragment block bIdx = kc*16 + nt holds B[k = kc*32 .. +31][n = nt*16 .. +15].
// Lane l (of 32): n = nt*16 + (l&15); K run = kc*32 + (l>=16 ? 16 : 0) + i,
// i = 0..15 stored contiguously (16 bf16 = 32B per lane).
// B[k][n] = W[n][k] (torch Linear: y = x @ W.T).
__global__ void k_prepW(const float* __restrict__ W, unsigned short* __restrict__ Wf) {
    int t    = blockIdx.x * blockDim.x + threadIdx.x;   // 65536 threads
    int bIdx = t >> 9;
    int r    = t & 511;
    int lane = r >> 4;
    int i    = r & 15;
    int kc   = bIdx >> 4;
    int nt   = bIdx & 15;
    int k    = kc * 32 + ((lane >= 16) ? 16 : 0) + i;
    int j    = nt * 16 + (lane & 15);
    Wf[t] = f32_to_bf16(W[j * HIDDEN + k]);
}

// ----------------------------------- scatter-add relu(E) into M_v (fp32 atomics)
__global__ void k_scatter(const float* __restrict__ E, const int* __restrict__ ei,
                          float* __restrict__ Mv) {
    long long t = (long long)blockIdx.x * blockDim.x + threadIdx.x;
    int edge = (int)(t >> 6);          // 64 threads per edge
    int q    = (int)(t & 63);          // float4 column
    if (edge >= N_EDGES) return;
    const float4 e4 = *reinterpret_cast<const float4*>(E + (size_t)edge * HIDDEN + q * 4);
    int dest = ei[N_EDGES + edge];     // edge_index row 1
    float* p = Mv + (size_t)dest * HIDDEN + q * 4;
    __hip_atomic_fetch_add(p + 0, fmaxf(e4.x, 0.f), __ATOMIC_RELAXED, __HIP_MEMORY_SCOPE_AGENT);
    __hip_atomic_fetch_add(p + 1, fmaxf(e4.y, 0.f), __ATOMIC_RELAXED, __HIP_MEMORY_SCOPE_AGENT);
    __hip_atomic_fetch_add(p + 2, fmaxf(e4.z, 0.f), __ATOMIC_RELAXED, __HIP_MEMORY_SCOPE_AGENT);
    __hip_atomic_fetch_add(p + 3, fmaxf(e4.w, 0.f), __ATOMIC_RELAXED, __HIP_MEMORY_SCOPE_AGENT);
}

// ---------------------------- fused gather + (M_v[src]-relu(E[rev])) @ W.T + b
#define TM 64            // edge rows per block
#define AS 264           // LDS row stride in ushorts (528B, 16B aligned, bank-skewed)

// FULL=true: block is entirely interior (no bounds checks anywhere).
template <bool FULL>
__global__ __launch_bounds__(256)
void k_gemm(const float* __restrict__ Mv, const float* __restrict__ E,
            const int* __restrict__ ei, const int* __restrict__ rev,
            const unsigned short* __restrict__ Wf, const float* __restrict__ bias,
            float* __restrict__ out, int blockBase) {
    __shared__ unsigned short sA[TM * AS];

    const int tid       = threadIdx.x;
    const int blockEdge = (blockBase + blockIdx.x) * TM;

    // Stage A tile: 64 rows x 256 cols of bf16(M_v[src[e]] - relu(E[rev[e]]))
    for (int it = 0; it < 16; ++it) {
        int lin = it * 256 + tid;          // 0..4095
        int r   = lin >> 6;                // row 0..63
        int c4  = lin & 63;                // float4 column
        int e   = blockEdge + r;
        float4 m4 = make_float4(0.f, 0.f, 0.f, 0.f);
        if (FULL || e < N_EDGES) {
            int s  = ei[e];                // edge_index row 0 (src)
            int rv = rev[e];
            const float4 mv = *reinterpret_cast<const float4*>(Mv + (size_t)s  * HIDDEN + c4 * 4);
            const float4 hv = *reinterpret_cast<const float4*>(E  + (size_t)rv * HIDDEN + c4 * 4);
            m4.x = mv.x - fmaxf(hv.x, 0.f);
            m4.y = mv.y - fmaxf(hv.y, 0.f);
            m4.z = mv.z - fmaxf(hv.z, 0.f);
            m4.w = mv.w - fmaxf(hv.w, 0.f);
        }
        unsigned short* d = &sA[r * AS + c4 * 4];
        d[0] = f32_to_bf16(m4.x);
        d[1] = f32_to_bf16(m4.y);
        d[2] = f32_to_bf16(m4.z);
        d[3] = f32_to_bf16(m4.w);
    }
    __syncthreads();

    const int wave   = tid >> 5;
    const int lane   = tid & 31;
    const int mt     = wave & 3;            // 4 M-tiles of 16 rows
    const int ntBase = (wave >> 2) * 8;     // waves 0-3: nt 0..7, waves 4-7: nt 8..15
    const int lrow   = lane & 15;
    const int hiHalf = lane >> 4;

    // Preload the 8 A K-fragments for this wave's 16-row strip from LDS.
    // A layout matches Wf packing: lane<16 -> K 0..15; lane>=16 -> K 16..31 (contiguous).
    union AF { uint4 q[2]; v16bf v; };
    AF a[8];
    const unsigned short* ap = &sA[(mt * 16 + lrow) * AS];
    #pragma unroll
    for (int kc = 0; kc < 8; ++kc) {
        int k0 = kc * 32 + hiHalf * 16;
        a[kc].q[0] = *reinterpret_cast<const uint4*>(ap + k0);
        a[kc].q[1] = *reinterpret_cast<const uint4*>(ap + k0 + 8);
    }

    #pragma unroll
    for (int ni = 0; ni < 8; ++ni) {
        const int nt = ntBase + ni;
        v8f acc = {};
        // Prefetch next N-tile's B column of fragments into cache hierarchy.
        __builtin_prefetch(Wf + ((size_t)(nt + 1) & 15) * 512 + (size_t)lane * 16, 0, 3);
        #pragma unroll
        for (int kc = 0; kc < 8; ++kc) {
            union { uint4 q[2]; v16bf v; } bf;
            const unsigned short* p = Wf + ((size_t)(kc * 16 + nt)) * 512 + (size_t)lane * 16;
            bf.q[0] = *reinterpret_cast<const uint4*>(p);
            bf.q[1] = *reinterpret_cast<const uint4*>(p + 8);
            acc = __builtin_amdgcn_wmma_f32_16x16x32_bf16(
                /*neg_a=*/false, a[kc].v, /*neg_b=*/false, bf.v,
                /*c_mod=*/(short)0, acc, /*reuse_a=*/false, /*reuse_b=*/false);
        }
        // D layout: lanes 0-15 -> N=lane, M=v; lanes 16-31 -> N=lane-16, M=v+8.
        const int n  = nt * 16 + lrow;
        const float bn = bias[n];
        #pragma unroll
        for (int v = 0; v < 8; ++v) {
            int m = mt * 16 + hiHalf * 8 + v;
            int e = blockEdge + m;
            if (FULL || e < N_EDGES)
                out[(size_t)e * HIDDEN + n] = acc[v] + bn;
        }
    }
}

// --------------------------------------------------------------------- launch
extern "C" void kernel_launch(void* const* d_in, const int* in_sizes, int n_in,
                              void* d_out, int out_size, void* d_ws, size_t ws_size,
                              hipStream_t stream) {
    const float* E    = (const float*)d_in[0];
    const int*   ei   = (const int*)d_in[1];   // [2, N_EDGES]: row0 src, row1 dest
    const int*   rev  = (const int*)d_in[2];
    const float* W    = (const float*)d_in[3];
    const float* bias = (const float*)d_in[4];
    float*       out  = (float*)d_out;
    (void)in_sizes; (void)n_in; (void)out_size; (void)ws_size;

    float*          Mv = (float*)d_ws;                                   // 51.2 MB
    unsigned short* Wf = (unsigned short*)((char*)d_ws +
                          (size_t)N_NODES * HIDDEN * sizeof(float));     // 128 KB

    const int n4 = N_NODES * HIDDEN / 4;
    k_zero   <<<(n4 + 255) / 256, 256, 0, stream>>>((float4*)Mv, n4);
    k_prepW  <<<(HIDDEN * HIDDEN) / 256, 256, 0, stream>>>(W, Wf);
    k_scatter<<<(N_EDGES * 64) / 256, 256, 0, stream>>>(E, ei, Mv);

    const int fullBlocks = N_EDGES / TM;       // 4687 fully-interior blocks
    const int tailBlocks = ((N_EDGES % TM) != 0) ? 1 : 0;
    k_gemm<true> <<<fullBlocks, 256, 0, stream>>>(Mv, E, ei, rev, Wf, bias, out, 0);
    if (tailBlocks)
        k_gemm<false><<<tailBlocks, 256, 0, stream>>>(Mv, E, ei, rev, Wf, bias, out, fullBlocks);
}
